// ModulatedDeformConvPack_76948634075498
// MI455X (gfx1250) — matrix operand order repaired and checked by reading, not compiled, not run on hardware
//
#include <hip/hip_runtime.h>

// ---------------------------------------------------------------------------
// DCNv2 for MI455X (gfx1250, wave32, WMMA).
//
// Shapes: B=2, C=64, H=W=128, O=64, 3x3 kernel (KK=9), stride=1, pad=1.
// P = Ho*Wo = 16384. K-dim of main GEMM = C*KK = 576 = 18 * 32.
//
// K ordering is ck = kk*64 + c (kk-major). The GEMM is invariant to any
// consistent permutation of K; kk-major makes the im2col writes fully
// contiguous (64 halves = 128 B per thread, stored as 8x b128).
//
// Stage 0: w_dcn f32 -> f16, permuted to [o][kk*64+c]            (73 KB)
// Stage 1: 27-ch offset/mask conv. offset == om[:, :18] + bias -> d_out tail.
//          mask = sigmoid(om[:, 18:27] + bias) -> ws.
// Stage 2: deformable bilinear im2col * mask -> col f16 [b][p][ck] (38 MB, L2)
// Stage 3: out[b,o,p] = sum_ck Wf16[o,ck]*col[b,p,ck] + bias via
//          v_wmma_f32_16x16x32_f16; each wave does 16x32 out (A reused
//          across two N-subtiles), 18 k-steps fully unrolled.
// ---------------------------------------------------------------------------

typedef __attribute__((ext_vector_type(16))) _Float16 v16h;
typedef __attribute__((ext_vector_type(8)))  _Float16 v8h;
typedef __attribute__((ext_vector_type(8)))  float    v8f;

#define BB   2
#define CC   64
#define HH   128
#define WW_  128
#define OO   64
#define KKN  9
#define PP   (HH * WW_)       // 16384
#define KDIM (CC * KKN)       // 576
#define KSTEPS (KDIM / 32)    // 18

// ---------------- Stage 0: weight f32 -> f16, permute K to kk-major --------
__global__ void cvt_w_f16(const float* __restrict__ w, _Float16* __restrict__ wf, int n) {
    int i = blockIdx.x * blockDim.x + threadIdx.x;   // destination index
    if (i >= n) return;
    const int o   = i / KDIM;
    const int rem = i - o * KDIM;       // = kk*64 + c
    const int kk  = rem >> 6;
    const int c   = rem & 63;
    wf[i] = (_Float16)w[(size_t)o * KDIM + c * KKN + kk];
}

// ---------------- Stage 1: offset/mask conv (direct, 27 out ch) ------------
__global__ __launch_bounds__(256) void om_conv(
    const float* __restrict__ xin,      // [B,C,H,W] input_offset
    const float* __restrict__ w,        // [27,C,3,3]
    const float* __restrict__ bias,     // [27]
    float* __restrict__ offset_out,     // [B,18,P]  (d_out tail, bias applied)
    float* __restrict__ mask_out)       // [B,9,P]   (ws, sigmoid applied)
{
    const int p  = blockIdx.x * blockDim.x + threadIdx.x;   // pixel
    const int oc = blockIdx.y;                              // 0..26
    const int b  = blockIdx.z;
    const int ho = p >> 7, wo = p & 127;

    const float* wptr = w + (size_t)oc * CC * 9;
    const float* xb   = xin + (size_t)b * CC * PP;

    float acc = 0.f;
    for (int c = 0; c < CC; ++c) {
        const float* xc = xb + (size_t)c * PP;
        const float* wc = wptr + c * 9;
#pragma unroll
        for (int dy = 0; dy < 3; ++dy) {
            const int iy = ho + dy - 1;
            if (iy < 0 || iy >= HH) continue;
#pragma unroll
            for (int dx = 0; dx < 3; ++dx) {
                const int ix = wo + dx - 1;
                if (ix < 0 || ix >= WW_) continue;
                acc = fmaf(wc[dy * 3 + dx], xc[iy * WW_ + ix], acc);
            }
        }
    }
    acc += bias[oc];
    if (oc < 18) {
        offset_out[((size_t)(b * 18 + oc)) * PP + p] = acc;
    } else {
        const float s = 1.f / (1.f + __expf(-acc));
        mask_out[((size_t)(b * 9 + (oc - 18))) * PP + p] = s;
    }
}

// ---------------- Stage 2: deformable bilinear im2col ----------------------
__global__ __launch_bounds__(256) void deform_im2col(
    const float* __restrict__ xin,      // [B,C,H,W] input_real
    const float* __restrict__ offs,     // [B,18,P] bias-applied offsets
    const float* __restrict__ mask,     // [B,9,P]
    _Float16* __restrict__ col)         // [B,P,KDIM], ck = kk*64 + c
{
    const int p  = blockIdx.x * blockDim.x + threadIdx.x;
    const int kk = blockIdx.y;          // 0..8
    const int b  = blockIdx.z;
    const int ho = p >> 7, wo = p & 127;

    const float offy = offs[((size_t)(b * 18 + 2 * kk)) * PP + p];
    const float offx = offs[((size_t)(b * 18 + 2 * kk + 1)) * PP + p];
    const float m    = mask[((size_t)(b * 9 + kk)) * PP + p];

    const float y = (float)(kk / 3 + ho - 1) + offy;
    const float x = (float)(kk % 3 + wo - 1) + offx;
    const float fy = floorf(y), fx = floorf(x);
    const float ly = y - fy, lx = x - fx;
    const float hy = 1.f - ly, hx = 1.f - lx;
    const int y0 = (int)fy, x0 = (int)fx;
    const int y1 = y0 + 1,  x1 = x0 + 1;

    const float vy0 = (y0 >= 0 && y0 < HH)  ? 1.f : 0.f;
    const float vy1 = (y1 >= 0 && y1 < HH)  ? 1.f : 0.f;
    const float vx0 = (x0 >= 0 && x0 < WW_) ? 1.f : 0.f;
    const float vx1 = (x1 >= 0 && x1 < WW_) ? 1.f : 0.f;

    // Fold mask into the bilinear weights (saves one mul per channel).
    const float w00 = hy * hx * vy0 * vx0 * m;
    const float w01 = hy * lx * vy0 * vx1 * m;
    const float w10 = ly * hx * vy1 * vx0 * m;
    const float w11 = ly * lx * vy1 * vx1 * m;

    const int cy0 = min(max(y0, 0), HH - 1),  cy1 = min(max(y1, 0), HH - 1);
    const int cx0 = min(max(x0, 0), WW_ - 1), cx1 = min(max(x1, 0), WW_ - 1);
    const int i00 = cy0 * WW_ + cx0, i01 = cy0 * WW_ + cx1;
    const int i10 = cy1 * WW_ + cx0, i11 = cy1 * WW_ + cx1;

    const float* xb = xin + (size_t)b * CC * PP;
    // 64 contiguous halves: base is 16B-aligned (1152*p + 128*kk bytes).
    _Float16* cptr = col + ((size_t)b * PP + p) * KDIM + kk * CC;

    for (int c0 = 0; c0 < CC; c0 += 8) {
        union { v8h h; float4 f; } pk;
#pragma unroll
        for (int j = 0; j < 8; ++j) {
            const float* xc = xb + (size_t)(c0 + j) * PP;
            const float v = w00 * xc[i00] + w01 * xc[i01]
                          + w10 * xc[i10] + w11 * xc[i11];
            pk.h[j] = (_Float16)v;
        }
        *(float4*)(cptr + c0) = pk.f;   // global_store_b128
    }
}

// ---------------- Stage 3: WMMA GEMM out = W * col + b ---------------------
// Block = 256 threads = 8 waves. Wave (mt,nt): mt = wave&3 -> M tile 16*mt,
// nt = wave>>2 -> 32-wide N tile (two 16x16 WMMAs sharing the A fragment).
// Block covers 64(M) x 64(N) of out[b][O][P].
__global__ __launch_bounds__(256) void dcn_wmma_gemm(
    const _Float16* __restrict__ wf16,  // [O, KDIM] row-major f16 (kk-major K)
    const _Float16* __restrict__ col,   // [B, P, KDIM] f16
    const float* __restrict__ bias,     // [O]
    float* __restrict__ out)            // [B, O, P]
{
    const int lane = threadIdx.x & 31;
    const int wave = threadIdx.x >> 5;
    const int m0   = (wave & 3) * 16;
    const int n0   = blockIdx.x * 64 + (wave >> 2) * 32;
    const int b    = blockIdx.y;
    const int kg   = lane >> 4;     // K half-group selector (CDNA5 16-bit layouts)
    const int mr   = lane & 15;     // A-row (M) and B-column (N) within tile

    // A fragment source: weight row (m0+mr), two 16B chunks per k-step.
    const _Float16* arow  = wf16 + (size_t)(m0 + mr) * KDIM;
    // B fragment sources: columns (n0+mr) and (n0+16+mr), 32B each per k-step.
    const _Float16* brow0 = col + ((size_t)b * PP + (size_t)(n0 + mr)) * KDIM;
    const _Float16* brow1 = brow0 + (size_t)16 * KDIM;

    v8f acc0 = {};
    v8f acc1 = {};
#pragma unroll
    for (int ks = 0; ks < KSTEPS; ++ks) {
        const int kbase = ks * 32;
        // A 16x32 f16: lanes 0-15 hold K = kbase+{0..15}; lanes 16-31
        // hold K = kbase+{16..31}  (contiguous order, matching B).
        union { v16h v; float4 f[2]; } a;
        a.f[0] = *(const float4*)(arow + kbase + kg * 16);
        a.f[1] = *(const float4*)(arow + kbase + kg * 16 + 8);
        // B 32x16 f16: lane n holds column n&15, K = kbase + (n>>4)*16 + 0..15.
        const v16h bf0 = *(const v16h*)(brow0 + kbase + kg * 16);
        const v16h bf1 = *(const v16h*)(brow1 + kbase + kg * 16);
        acc0 = __builtin_amdgcn_wmma_f32_16x16x32_f16(
            false, a.v, false, bf0, (short)0, acc0, false, false);
        acc1 = __builtin_amdgcn_wmma_f32_16x16x32_f16(
            false, a.v, false, bf1, (short)0, acc1, false, false);
    }

    // C/D layout: VGPR r, lanes 0-15 -> M=r, lanes 16-31 -> M=r+8; N = lane&15.
    const int ncol0 = n0 + mr;
#pragma unroll
    for (int r = 0; r < 8; ++r) {
        const int mrow = m0 + r + (kg ? 8 : 0);
        const float bv = bias[mrow];
        float* orow = out + ((size_t)(b * OO + mrow)) * PP;
        orow[ncol0]      = acc0[r] + bv;
        orow[ncol0 + 16] = acc1[r] + bv;
    }
}

// ---------------------------------------------------------------------------
extern "C" void kernel_launch(void* const* d_in, const int* in_sizes, int n_in,
                              void* d_out, int out_size, void* d_ws, size_t ws_size,
                              hipStream_t stream) {
    const float* in_off  = (const float*)d_in[0];   // input_offset [B,C,H,W]
    const float* in_real = (const float*)d_in[1];   // input_real   [B,C,H,W]
    const float* w_om    = (const float*)d_in[2];   // [27,C,3,3]
    const float* b_om    = (const float*)d_in[3];   // [27]
    const float* w_dcn   = (const float*)d_in[4];   // [O,C,3,3]
    const float* b_dcn   = (const float*)d_in[5];   // [O]

    float* out        = (float*)d_out;                       // [B,O,P]
    float* offset_out = out + (size_t)BB * OO * PP;          // [B,18,P]

    // ws layout: mask (1.2 MB) | wf16 @2MB (73 KB) | col f16 @4MB (38 MB)
    char* ws = (char*)d_ws;
    float*    mask = (float*)ws;
    _Float16* wf16 = (_Float16*)(ws + ((size_t)2 << 20));
    _Float16* col  = (_Float16*)(ws + ((size_t)4 << 20));

    hipLaunchKernelGGL(cvt_w_f16, dim3((OO * KDIM + 255) / 256), dim3(256), 0, stream,
                       w_dcn, wf16, OO * KDIM);

    hipLaunchKernelGGL(om_conv, dim3(PP / 256, 27, BB), dim3(256), 0, stream,
                       in_off, w_om, b_om, offset_out, mask);

    hipLaunchKernelGGL(deform_im2col, dim3(PP / 256, KKN, BB), dim3(256), 0, stream,
                       in_real, offset_out, mask, col);

    hipLaunchKernelGGL(dcn_wmma_gemm, dim3(PP / 64, BB), dim3(256), 0, stream,
                       wf16, col, b_dcn, out);
}
